// RSSM_35734127903500
// MI455X (gfx1250) — matrix operand order repaired and checked by reading, not compiled, not run on hardware
//
#include <hip/hip_runtime.h>
#include <stdint.h>

#define BB   1024
#define TT   16
#define AA   6
#define SS   32
#define KK   32
#define LL   1024
#define HH   2048
#define GRU_IN  (LL + HH)   // 3072
#define GRU_OUT (3 * HH)    // 6144
#define FEAT (LL + HH)      // 3072 output features per (b,t)

typedef __attribute__((ext_vector_type(16))) __bf16 v16bf;
typedef __attribute__((ext_vector_type(8)))  float  v8f;

union FragU { v16bf v; unsigned int u[8]; uint4 q[2]; };

__device__ __forceinline__ unsigned short f32_to_bf16(float f) {
  union { float f; unsigned int u; } a; a.f = f;
  unsigned int r = a.u + 0x7fffu + ((a.u >> 16) & 1u);  // round-to-nearest-even
  return (unsigned short)(r >> 16);
}
__device__ __forceinline__ float sigmoidf_(float x) { return 1.0f / (1.0f + __expf(-x)); }

// gfx1250 async copy global->LDS, 16B per lane, tracked by ASYNCcnt
__device__ __forceinline__ void async_ld_b128(unsigned lds_addr, unsigned long long gaddr) {
  asm volatile("global_load_async_to_lds_b128 %0, %1, off" :: "v"(lds_addr), "v"(gaddr) : "memory");
}
__device__ __forceinline__ void wait_async0() {
  asm volatile("s_wait_asynccnt 0x0" ::: "memory");
}

// ------- weight convert f32 -> bf16 with WMMA-fragment swizzle -------------
// Block per (kt = k/32, nb = n/16): 32 lanes x 16 bf16 (1KB), lane-major.
__global__ void k_swz(const float* __restrict__ src, unsigned short* __restrict__ dst,
                      int Kd, int Nd) {
  int i = blockIdx.x * blockDim.x + threadIdx.x;
  if (i >= Kd * Nd) return;
  int k = i / Nd, n = i % Nd;
  int kt = k >> 5, kr = k & 31;
  int half = (kr >> 3) & 1;
  int kb = kr - 8 * half;                       // {0..7} u {16..23}
  int ii = (kb < 8) ? (kb >> 1) : (4 + ((kb - 16) >> 1));
  int parity = kb & 1;
  int nb = n >> 4, nn = n & 15;
  int lane = half * 16 + nn;
  size_t dstIdx = (((size_t)kt * (Nd >> 4) + nb) * 32 + lane) * 16 + 2 * ii + parity;
  dst[dstIdx] = f32_to_bf16(src[(size_t)k * Nd + n]);
}

// ---------------- state init: xd=0, deter=0, idx=-1 ----------------
__global__ void k_init(unsigned short* __restrict__ xd, float* __restrict__ deter, int* __restrict__ idx) {
  int i = blockIdx.x * blockDim.x + threadIdx.x;
  if (i < BB * GRU_IN) xd[i] = 0;
  if (i < BB * HH)     deter[i] = 0.0f;
  if (i < BB * SS)     idx[i] = -1;
}

// ---------------- bf16 WMMA GEMM: C[M,N] = A[M,KDIM] @ Wswz --------------
// 256 thr = 8 waves (4 in M x 2 in N); wave tile = (MI*16) x 64 -> MI x 4 wmma.
// MI=4: block 256x128, 16 wmma per 8 frag-loads per wave (big GEMM).
// MI=2: block 128x128 (small GEMMs, keeps 64 blocks in flight).
// Double-buffered LDS fed by global_load_async_to_lds_b128; one fb live at a time.
#define TN 128
#define TK 32
#define LDA_S 40    // A LDS row stride: 80B, 16B aligned, bank-skewed

template<int KDIM, int N, int LDA, int MI>
__global__ __launch_bounds__(256)
void k_gemm(const unsigned short* __restrict__ A,
            const unsigned short* __restrict__ Wsw,
            float* __restrict__ C)
{
  constexpr int BM    = MI * 64;        // block M tile (128 or 256)
  constexpr int ELEMS = BM / 8;         // bf16 per thread for the A tile copy
  constexpr int TPR   = 32 / ELEMS;     // threads per A row (2 or 1)
  constexpr int CH    = ELEMS / 8;      // 16B chunks per thread (2 or 4)

  __shared__ unsigned short As[2][BM * LDA_S];
  __shared__ unsigned short Bq[2][TK * TN];     // fragment-order, 8KB each

  const int tid  = threadIdx.x;
  const int lane = tid & 31;
  const int wave = tid >> 5;
  const int waveM = wave >> 1;     // 0..3
  const int waveN = wave & 1;      // 0..1
  const int m16  = lane & 15;
  const int half = lane >> 4;

  const int tileM = blockIdx.y * BM;
  const int tileN = blockIdx.x * TN;

  const int ar = tid / TPR;
  const int ac = (tid % TPR) * ELEMS;

  const unsigned aL[2] = { (unsigned)(size_t)&As[0][ar * LDA_S + ac],
                           (unsigned)(size_t)&As[1][ar * LDA_S + ac] };
  const unsigned bL[2] = { (unsigned)(size_t)&Bq[0][tid * 16],
                           (unsigned)(size_t)&Bq[1][tid * 16] };
  const unsigned short* gA0 = A + (size_t)(tileM + ar) * LDA + ac;
  const unsigned short* gB0 = Wsw + (size_t)tileN * 32 + tid * 16;

  auto issue = [&](int buf, int k0) {
    unsigned long long ga = (unsigned long long)(size_t)(gA0 + k0);
    unsigned long long gb = (unsigned long long)(size_t)(gB0 + (size_t)k0 * N);
    #pragma unroll
    for (int c = 0; c < CH; c++) async_ld_b128(aL[buf] + 16 * c, ga + 16 * c);
    async_ld_b128(bL[buf],      gb);
    async_ld_b128(bL[buf] + 16, gb + 16);
  };

  v8f acc[MI][4];
  const v8f vzero = {0.f, 0.f, 0.f, 0.f, 0.f, 0.f, 0.f, 0.f};
  #pragma unroll
  for (int mi = 0; mi < MI; mi++)
    #pragma unroll
    for (int ni = 0; ni < 4; ni++) acc[mi][ni] = vzero;

  issue(0, 0);
  for (int k0 = 0; k0 < KDIM; k0 += TK) {
    const int cur = (k0 >> 5) & 1;
    wait_async0();           // own async loads for buffer `cur` done
    __syncthreads();         // everyone's `cur` ready; prior reads of `cur^1` done
    if (k0 + TK < KDIM) issue(cur ^ 1, k0 + TK);   // overlap next tile with compute

    const unsigned short* AsC = As[cur];
    const unsigned short* BqC = Bq[cur];

    FragU fa[MI];
    #pragma unroll
    for (int mi = 0; mi < MI; mi++) {
      int rb = (waveM * (MI * 16) + mi * 16 + m16) * LDA_S + 8 * half;
      fa[mi].q[0] = *(const uint4*)&AsC[rb];        // K pairs i=0..3
      fa[mi].q[1] = *(const uint4*)&AsC[rb + 16];   // K pairs i=4..7
    }
    #pragma unroll
    for (int ni = 0; ni < 4; ni++) {
      FragU fb;
      int base = ((waveN * 4 + ni) * 32 + lane) * 16;
      fb.q[0] = *(const uint4*)&BqC[base];
      fb.q[1] = *(const uint4*)&BqC[base + 8];
      #pragma unroll
      for (int mi = 0; mi < MI; mi++)
        acc[mi][ni] = __builtin_amdgcn_wmma_f32_16x16x32_bf16(
            false, fa[mi].v, false, fb.v, (short)0, acc[mi][ni], false, false);
    }
  }

  // C/D layout: VGPR v -> M = v + 8*half, N = lane&15 ; N compile-time -> imm offsets
  #pragma unroll
  for (int mi = 0; mi < MI; mi++) {
    int rbase = tileM + waveM * (MI * 16) + mi * 16 + 8 * half;
    #pragma unroll
    for (int ni = 0; ni < 4; ni++) {
      int col = tileN + waveN * 64 + ni * 16 + m16;
      float* cp = C + (size_t)rbase * N + col;
      #pragma unroll
      for (int v = 0; v < 8; v++) cp[(size_t)v * N] = acc[mi][ni][v];
    }
  }
}

// ---------------- stage 1: one-hot gather @ W_pre + action + LN + silu -> xd[:, :1024] (bf16)
__global__ __launch_bounds__(256)
void k_pre(const int* __restrict__ idx, const float* __restrict__ action, int t,
           const float* __restrict__ Wpre, const float* __restrict__ g, const float* __restrict__ bv,
           unsigned short* __restrict__ xd)
{
  __shared__ float red[256];
  const int b = blockIdx.x, tid = threadIdx.x;

  float an[AA];
  const float* ap = action + ((size_t)b * TT + t) * AA;
  #pragma unroll
  for (int j = 0; j < AA; j++) { float a = ap[j]; float m = fabsf(a); an[j] = a / (m > 1.0f ? m : 1.0f); }

  float acc[4] = {0.f, 0.f, 0.f, 0.f};
  for (int s = 0; s < SS; s++) {
    int r = idx[b * SS + s];
    if (r >= 0) {
      const float* wr = Wpre + (size_t)(s * KK + r) * LL;
      #pragma unroll
      for (int p = 0; p < 4; p++) acc[p] += wr[tid + 256 * p];
    }
  }
  #pragma unroll
  for (int j = 0; j < AA; j++) {
    const float* wr = Wpre + (size_t)(SS * KK + j) * LL;
    #pragma unroll
    for (int p = 0; p < 4; p++) acc[p] += an[j] * wr[tid + 256 * p];
  }

  float s1 = 0.f, s2 = 0.f;
  #pragma unroll
  for (int p = 0; p < 4; p++) { s1 += acc[p]; s2 += acc[p] * acc[p]; }
  red[tid] = s1; __syncthreads();
  for (int off = 128; off > 0; off >>= 1) { if (tid < off) red[tid] += red[tid + off]; __syncthreads(); }
  float mean = red[0] / (float)LL; __syncthreads();
  red[tid] = s2; __syncthreads();
  for (int off = 128; off > 0; off >>= 1) { if (tid < off) red[tid] += red[tid + off]; __syncthreads(); }
  float var = red[0] / (float)LL - mean * mean;
  float rstd = rsqrtf(var + 1e-3f);
  #pragma unroll
  for (int p = 0; p < 4; p++) {
    int j = tid + 256 * p;
    float z = (acc[p] - mean) * rstd * g[j] + bv[j];
    float x = z * sigmoidf_(z);
    xd[(size_t)b * GRU_IN + j] = f32_to_bf16(x);
  }
}

// ---------------- stage 2b: LN(6144) + GRU gates -> deter, xd[:,1024:], feat out
__global__ __launch_bounds__(256)
void k_gru(const float* __restrict__ raw, const float* __restrict__ g, const float* __restrict__ bg,
           float* __restrict__ deter, unsigned short* __restrict__ xd, float* __restrict__ out, int t)
{
  __shared__ float red[256];
  const int b = blockIdx.x, tid = threadIdx.x;
  const float* rr = raw + (size_t)b * GRU_OUT;

  float s1 = 0.f, s2 = 0.f;
  #pragma unroll
  for (int p = 0; p < 24; p++) { float v = rr[tid + 256 * p]; s1 += v; s2 += v * v; }
  red[tid] = s1; __syncthreads();
  for (int off = 128; off > 0; off >>= 1) { if (tid < off) red[tid] += red[tid + off]; __syncthreads(); }
  float mean = red[0] / (float)GRU_OUT; __syncthreads();
  red[tid] = s2; __syncthreads();
  for (int off = 128; off > 0; off >>= 1) { if (tid < off) red[tid] += red[tid + off]; __syncthreads(); }
  float var = red[0] / (float)GRU_OUT - mean * mean;
  float rstd = rsqrtf(var + 1e-3f);

  #pragma unroll
  for (int q = 0; q < 8; q++) {
    int h = tid + 256 * q;
    float nr = (rr[h]        - mean) * rstd * g[h]        + bg[h];
    float nc = (rr[2048 + h] - mean) * rstd * g[2048 + h] + bg[2048 + h];
    float nu = (rr[4096 + h] - mean) * rstd * g[4096 + h] + bg[4096 + h];
    float r = sigmoidf_(nr);
    float c = tanhf(r * nc);
    float u = sigmoidf_(nu - 1.0f);
    float dn = u * c + (1.0f - u) * deter[(size_t)b * HH + h];
    deter[(size_t)b * HH + h] = dn;
    xd[(size_t)b * GRU_IN + LL + h] = f32_to_bf16(dn);
    out[((size_t)b * TT + t) * FEAT + LL + h] = dn;
  }
}

// ---------------- stage 3b: LN(1024) + silu -> y (bf16)
__global__ __launch_bounds__(256)
void k_lnsilu(const float* __restrict__ raw, const float* __restrict__ g, const float* __restrict__ bv,
              unsigned short* __restrict__ y)
{
  __shared__ float red[256];
  const int b = blockIdx.x, tid = threadIdx.x;
  const float* rr = raw + (size_t)b * LL;
  float acc[4];
  #pragma unroll
  for (int p = 0; p < 4; p++) acc[p] = rr[tid + 256 * p];
  float s1 = 0.f, s2 = 0.f;
  #pragma unroll
  for (int p = 0; p < 4; p++) { s1 += acc[p]; s2 += acc[p] * acc[p]; }
  red[tid] = s1; __syncthreads();
  for (int off = 128; off > 0; off >>= 1) { if (tid < off) red[tid] += red[tid + off]; __syncthreads(); }
  float mean = red[0] / (float)LL; __syncthreads();
  red[tid] = s2; __syncthreads();
  for (int off = 128; off > 0; off >>= 1) { if (tid < off) red[tid] += red[tid + off]; __syncthreads(); }
  float var = red[0] / (float)LL - mean * mean;
  float rstd = rsqrtf(var + 1e-3f);
  #pragma unroll
  for (int p = 0; p < 4; p++) {
    int j = tid + 256 * p;
    float z = (acc[p] - mean) * rstd * g[j] + bv[j];
    float x = z * sigmoidf_(z);
    y[(size_t)b * LL + j] = f32_to_bf16(x);
  }
}

// ---------------- stage 4b: +bias, per-32 argmax -> idx, one-hot -> feat out
__global__ void k_stoch(const float* __restrict__ logit, const float* __restrict__ bias,
                        int* __restrict__ idx, float* __restrict__ out, int t)
{
  const int b = blockIdx.x;
  const int s = threadIdx.x;  // 32 threads = 1 wave, one group each
  const float* lp = logit + (size_t)b * LL + s * KK;
  const float* bp = bias + s * KK;
  float best = lp[0] + bp[0];
  int bk = 0;
  for (int k = 1; k < KK; k++) {
    float v = lp[k] + bp[k];
    if (v > best) { best = v; bk = k; }   // first-max tie-break (matches jnp.argmax)
  }
  idx[b * SS + s] = bk;
  float* op = out + ((size_t)b * TT + t) * FEAT + s * KK;
  for (int k = 0; k < KK; k++) op[k] = (k == bk) ? 1.0f : 0.0f;
}

extern "C" void kernel_launch(void* const* d_in, const int* in_sizes, int n_in,
                              void* d_out, int out_size, void* d_ws, size_t ws_size,
                              hipStream_t stream) {
  (void)in_sizes; (void)n_in; (void)out_size; (void)ws_size;
  const float* action = (const float*)d_in[0];
  const float* W_pre  = (const float*)d_in[1];
  const float* g_pre  = (const float*)d_in[2];
  const float* b_pre  = (const float*)d_in[3];
  const float* W_gru  = (const float*)d_in[4];
  const float* g_gru  = (const float*)d_in[5];
  const float* b_gru  = (const float*)d_in[6];
  const float* W_tr   = (const float*)d_in[7];
  const float* g_tr   = (const float*)d_in[8];
  const float* b_tr   = (const float*)d_in[9];
  const float* W_ims  = (const float*)d_in[10];
  const float* b_ims  = (const float*)d_in[11];
  float* out = (float*)d_out;

  char* ws = (char*)d_ws;
  size_t off = 0;
  auto alloc = [&](size_t bytes) -> char* {
    char* p = ws + off;
    off = (off + bytes + 255) & ~(size_t)255;
    return p;
  };
  unsigned short* Wg   = (unsigned short*)alloc((size_t)GRU_IN * GRU_OUT * 2);  // swizzled
  unsigned short* Wt   = (unsigned short*)alloc((size_t)HH * LL * 2);           // swizzled
  unsigned short* Wi   = (unsigned short*)alloc((size_t)LL * LL * 2);           // swizzled
  unsigned short* xd   = (unsigned short*)alloc((size_t)BB * GRU_IN * 2);
  unsigned short* yb   = (unsigned short*)alloc((size_t)BB * LL * 2);
  float* deter = (float*)alloc((size_t)BB * HH * 4);
  float* rawg  = (float*)alloc((size_t)BB * GRU_OUT * 4);
  float* rawt  = (float*)alloc((size_t)BB * LL * 4);
  float* logit = (float*)alloc((size_t)BB * LL * 4);
  int*   idx   = (int*)alloc((size_t)BB * SS * 4);

  int n;
  n = GRU_IN * GRU_OUT; k_swz<<<(n + 255) / 256, 256, 0, stream>>>(W_gru, Wg, GRU_IN, GRU_OUT);
  n = HH * LL;          k_swz<<<(n + 255) / 256, 256, 0, stream>>>(W_tr, Wt, HH, LL);
  n = LL * LL;          k_swz<<<(n + 255) / 256, 256, 0, stream>>>(W_ims, Wi, LL, LL);
  n = BB * GRU_IN;      k_init<<<(n + 255) / 256, 256, 0, stream>>>(xd, deter, idx);

  for (int t = 0; t < TT; t++) {
    // x = silu(LN(gather(W_pre, idx) + a_norm part))
    k_pre<<<BB, 256, 0, stream>>>(idx, action, t, W_pre, g_pre, b_pre, xd);
    // raw_gru = [x | deter] @ W_gru   (1024 x 3072 x 6144), 256x128 tiles
    k_gemm<GRU_IN, GRU_OUT, GRU_IN, 4>
        <<<dim3(GRU_OUT / TN, BB / 256), 256, 0, stream>>>(xd, Wg, rawg);
    // LN + gates -> deter (fp32 + bf16 copy into xd) + feat output
    k_gru<<<BB, 256, 0, stream>>>(rawg, g_gru, b_gru, deter, xd, out, t);
    // raw_tr = deter @ W_tr           (1024 x 2048 x 1024), 128x128 tiles
    k_gemm<HH, LL, GRU_IN, 2>
        <<<dim3(LL / TN, BB / 128), 256, 0, stream>>>(xd + LL, Wt, rawt);
    // y = silu(LN(raw_tr))
    k_lnsilu<<<BB, 256, 0, stream>>>(rawt, g_tr, b_tr, yb);
    // logit = y @ W_ims               (1024 x 1024 x 1024), 128x128 tiles
    k_gemm<LL, LL, LL, 2>
        <<<dim3(LL / TN, BB / 128), 256, 0, stream>>>(yb, Wi, logit);
    // argmax per 32-group -> idx, one-hot -> feat output
    k_stoch<<<BB, 32, 0, stream>>>(logit, b_ims, idx, out, t);
  }
}
